// GraphTransformer_6339371729769
// MI455X (gfx1250) — matrix operand order repaired and checked by reading, compiled, not run on hardware
//
#include <hip/hip_runtime.h>
#include <hip/hip_bf16.h>

typedef __attribute__((ext_vector_type(16))) __bf16 v16bf;
typedef __attribute__((ext_vector_type(8)))  __bf16 v8bf;
typedef __attribute__((ext_vector_type(8)))  float  v8f;
typedef __attribute__((ext_vector_type(4)))  int    v4i;

#define NN   8192
#define CC   128
#define KNN  7

#define GLOBAL_AS __attribute__((address_space(1)))
#define LDS_AS    __attribute__((address_space(3)))

// ---------------- CDNA5 async copy: global -> LDS (ASYNCcnt tracked) --------
__device__ __forceinline__ void async_b128(const __bf16* g, __bf16* l) {
#if __has_builtin(__builtin_amdgcn_global_load_async_to_lds_b128)
  __builtin_amdgcn_global_load_async_to_lds_b128(
      (GLOBAL_AS v4i*)g, (LDS_AS v4i*)l, 0, 0);
#else
  asm volatile("global_load_async_to_lds_b128 %0, %1, off"
               :: "v"((unsigned)(unsigned long long)(void*)l), "v"(g) : "memory");
#endif
}
__device__ __forceinline__ void wait_async_le32() {
#if __has_builtin(__builtin_amdgcn_s_wait_asynccnt)
  __builtin_amdgcn_s_wait_asynccnt(32);
#else
  asm volatile("s_wait_asynccnt 32" ::: "memory");
#endif
}
__device__ __forceinline__ void wait_async_0() {
#if __has_builtin(__builtin_amdgcn_s_wait_asynccnt)
  __builtin_amdgcn_s_wait_asynccnt(0);
#else
  asm volatile("s_wait_asynccnt 0" ::: "memory");
#endif
}

// ---------------- fragment helpers ------------------------------------------
// build v16bf from two contiguous 8-element (16B) runs
__device__ __forceinline__ v16bf frag2x8(const __bf16* p0, const __bf16* p1) {
  v8bf lo = *(const v8bf*)p0;
  v8bf hi = *(const v8bf*)p1;
  v16bf r;
#pragma unroll
  for (int e = 0; e < 8; ++e) { r[e] = lo[e]; r[e + 8] = hi[e]; }
  return r;
}
// A-frag 16x32: lane: row=m0+L%16; elems 0-15 at k0+g*16+e
__device__ __forceinline__ v16bf load_a_bf(const __bf16* p, int ld, int m0, int k0, int lane) {
  const __bf16* b = p + (size_t)(m0 + (lane & 15)) * ld + k0 + ((lane >> 4) << 4);
  return frag2x8(b, b + 8);
}
// fp32-source A/B frags (used only by the small projection GEMMs)
__device__ __forceinline__ v16bf load_a_f32(const float* __restrict__ p, int ld, int m0, int k0) {
  int lane = threadIdx.x & 31;
  int r = m0 + (lane & 15), g = (lane >> 4) << 4;
  v16bf a;
#pragma unroll
  for (int e = 0; e < 16; ++e)
    a[e] = (__bf16)p[(size_t)r * ld + k0 + g + e];
  return a;
}
__device__ __forceinline__ v16bf load_b_f32(const float* __restrict__ p, int ld, int k0, int n0) {
  int lane = threadIdx.x & 31;
  int n = n0 + (lane & 15), g = (lane >> 4) << 4;
  v16bf b;
#pragma unroll
  for (int e = 0; e < 16; ++e)
    b[e] = (__bf16)p[(size_t)(k0 + g + e) * ld + n];
  return b;
}
__device__ __forceinline__ v8f wmma_bf16(v16bf a, v16bf b, v8f c) {
  return __builtin_amdgcn_wmma_f32_16x16x32_bf16(false, a, false, b, (short)0, c, false, false);
}

// ---------------- GEMM (+bias) with optional bf16 rm / bf16 transposed out --
__global__ __launch_bounds__(32) void gemm_bias(const float* __restrict__ X,
                                                const float* __restrict__ W,
                                                const float* __restrict__ bias,
                                                float* __restrict__ outf,
                                                __bf16* __restrict__ out_rm,
                                                __bf16* __restrict__ out_tr,
                                                int K, int cout) {
  int m0 = blockIdx.x * 16, n0 = blockIdx.y * 16;
  v8f c = {0.f,0.f,0.f,0.f,0.f,0.f,0.f,0.f};
  for (int k0 = 0; k0 < K; k0 += 32)
    c = wmma_bf16(load_a_f32(X, K, m0, k0), load_b_f32(W, cout, k0, n0), c);
  int lane = threadIdx.x & 31;
  int col  = n0 + (lane & 15);
  int half = (lane >> 4) << 3;
  float bv = bias ? bias[col] : 0.f;
#pragma unroll
  for (int j = 0; j < 8; ++j) {
    size_t row = (size_t)(m0 + j + half);
    float v = c[j] + bv;
    if (outf)   outf[row * cout + col] = v;
    if (out_rm) out_rm[row * cout + col] = (__bf16)v;
    if (out_tr) out_tr[(size_t)col * NN + row] = (__bf16)v;   // [cout][NN]
  }
}

// ---------------- Flash attention: async LDS double-buffered K/V ------------
__global__ __launch_bounds__(32) void attn(const __bf16* __restrict__ Qbf,
                                           const __bf16* __restrict__ Kbf,
                                           const __bf16* __restrict__ Vt,
                                           const float* __restrict__ Skip,
                                           float* __restrict__ H) {
  __shared__ __align__(16) __bf16 ldsK[2][32 * CC];   // [key-in-chunk][channel]
  __shared__ __align__(16) __bf16 ldsV[2][CC * 32];   // [channel][key-in-chunk]
  __shared__ __align__(16) __bf16 pbuf[16 * 32];
  const float scale = 0.08838834764831845f;           // 1/sqrt(128)
  int m0   = blockIdx.x * 16;
  int lane = threadIdx.x & 31;
  int half = (lane >> 4) << 3;

  v16bf qa[4];
#pragma unroll
  for (int kk = 0; kk < 4; ++kk) qa[kk] = load_a_bf(Qbf, CC, m0, kk * 32, lane);

  v8f o[8];
  float mrun[8], lrun[8];
#pragma unroll
  for (int f = 0; f < 8; ++f) o[f] = (v8f){0.f,0.f,0.f,0.f,0.f,0.f,0.f,0.f};
#pragma unroll
  for (int j = 0; j < 8; ++j) { mrun[j] = -1.0e30f; lrun[j] = 0.f; }

  // issue one chunk's 32 async b128 copies (8KB K + 8KB V)
  auto issue = [&](int c0, int buf) {
    const __bf16* gk = Kbf + (size_t)c0 * CC;          // contiguous 8KB
#pragma unroll
    for (int i = 0; i < 16; ++i) {
      int off = (i * 32 + lane) * 8;
      async_b128(gk + off, &ldsK[buf][off]);
    }
#pragma unroll
    for (int i = 0; i < 16; ++i) {                     // Vt rows, 64B each
      int row = i * 8 + (lane >> 2);
      int seg = (lane & 3) * 8;
      async_b128(Vt + (size_t)row * NN + c0 + seg, &ldsV[buf][row * 32 + seg]);
    }
  };

  issue(0, 0);
  const int NCH = NN / 32;
  for (int ch = 0; ch < NCH; ++ch) {
    int cur = ch & 1, c0 = ch * 32;
    bool more = (ch + 1) < NCH;
    if (more) issue(c0 + 32, cur ^ 1);                 // prefetch next chunk
    if (more) wait_async_le32(); else wait_async_0();  // current chunk resident
    __syncthreads();

    // scores S = Q(16x128) @ K^T(128x32), two 16x16 tiles from LDS
    v8f s0 = (v8f){0.f,0.f,0.f,0.f,0.f,0.f,0.f,0.f};
    v8f s1 = s0;
#pragma unroll
    for (int kk = 0; kk < 4; ++kk) {
      const __bf16* b0 = &ldsK[cur][(lane & 15) * CC + kk * 32 + ((lane >> 4) << 4)];
      const __bf16* b1 = b0 + 16 * CC;
      s0 = wmma_bf16(qa[kk], frag2x8(b0, b0 + 8), s0);
      s1 = wmma_bf16(qa[kk], frag2x8(b1, b1 + 8), s1);
    }
    // online softmax (rows per-register, cols across 16 lanes)
#pragma unroll
    for (int j = 0; j < 8; ++j) {
      int row  = m0 + j + half;
      int col0 = c0 + (lane & 15);
      float a0 = s0[j] * scale, a1 = s1[j] * scale;
      if (col0 == row)      a0 = -1.0e30f;
      if (col0 + 16 == row) a1 = -1.0e30f;
      float mx = fmaxf(a0, a1);
#pragma unroll
      for (int d = 1; d < 16; d <<= 1) mx = fmaxf(mx, __shfl_xor(mx, d, 32));
      float mnew  = fmaxf(mrun[j], mx);
      float alpha = __expf(mrun[j] - mnew);
      float p0 = __expf(a0 - mnew), p1 = __expf(a1 - mnew);
      float ps = p0 + p1;
#pragma unroll
      for (int d = 1; d < 16; d <<= 1) ps += __shfl_xor(ps, d, 32);
      lrun[j] = lrun[j] * alpha + ps;
      mrun[j] = mnew;
#pragma unroll
      for (int f = 0; f < 8; ++f) o[f][j] *= alpha;
      pbuf[(j + half) * 32 + (lane & 15)]      = (__bf16)p0;
      pbuf[(j + half) * 32 + (lane & 15) + 16] = (__bf16)p1;
    }
    __syncthreads();
    // O += P(16x32) @ V(32x128); P reloaded in A-layout, V from LDS (tr layout)
    {
      const __bf16* pb = &pbuf[(lane & 15) * 32 + ((lane >> 4) << 4)];
      v16bf pa = frag2x8(pb, pb + 8);
#pragma unroll
      for (int f = 0; f < 8; ++f) {
        const __bf16* vb = &ldsV[cur][(f * 16 + (lane & 15)) * 32 + ((lane >> 4) << 4)];
        o[f] = wmma_bf16(pa, frag2x8(vb, vb + 8), o[f]);
      }
    }
    __syncthreads();
  }
  // epilogue: normalize, add skip projection, store h (fp32)
  int colb = lane & 15;
#pragma unroll
  for (int f = 0; f < 8; ++f) {
    int col = f * 16 + colb;
#pragma unroll
    for (int j = 0; j < 8; ++j) {
      size_t row = (size_t)(m0 + j + half);
      H[row * CC + col] = o[f][j] / lrun[j] + Skip[row * CC + col];
    }
  }
}

// ---------------- KNN -------------------------------------------------------
__global__ void sqnorm(const float* __restrict__ H, float* __restrict__ sq) {
  int i = blockIdx.x * blockDim.x + threadIdx.x;
  if (i >= NN) return;
  float s = 0.f;
  for (int c = 0; c < CC; ++c) { float v = H[(size_t)i * CC + c]; s = fmaf(v, v, s); }
  sq[i] = s;
}

__global__ __launch_bounds__(128) void knn(const float* __restrict__ H,
                                           const float* __restrict__ sq,
                                           int* __restrict__ idx) {
  __shared__ float hrow[CC];
  __shared__ float cd[128 * KNN];
  __shared__ int   ci[128 * KNN];
  int i = blockIdx.x, t = threadIdx.x;
  hrow[t] = H[(size_t)i * CC + t];
  __syncthreads();
  float bd[KNN]; int bi[KNN];
#pragma unroll
  for (int q = 0; q < KNN; ++q) { bd[q] = 1.0e30f; bi[q] = 0; }
  float sqi = sq[i];
  for (int j = t; j < NN; j += 128) {
    __builtin_prefetch(&H[(size_t)(j + 128) * CC], 0, 1);
    if (j == i) continue;
    float dot = 0.f;
    for (int c = 0; c < CC; ++c) dot = fmaf(hrow[c], H[(size_t)j * CC + c], dot);
    float d2 = sqi + sq[j] - 2.f * dot;
    if (d2 < bd[KNN - 1]) {
      bd[KNN - 1] = d2; bi[KNN - 1] = j;
#pragma unroll
      for (int q = KNN - 1; q > 0; --q)
        if (bd[q] < bd[q - 1]) {
          float td = bd[q]; bd[q] = bd[q - 1]; bd[q - 1] = td;
          int   ti = bi[q]; bi[q] = bi[q - 1]; bi[q - 1] = ti;
        }
    }
  }
#pragma unroll
  for (int q = 0; q < KNN; ++q) { cd[t * KNN + q] = bd[q]; ci[t * KNN + q] = bi[q]; }
  __syncthreads();
  if (t == 0) {
    for (int s = 0; s < KNN; ++s) {
      float best = 1.0e30f; int bj = 0, bp = 0;
      for (int p = 0; p < 128 * KNN; ++p)
        if (cd[p] < best) { best = cd[p]; bj = ci[p]; bp = p; }
      idx[(size_t)i * KNN + s] = bj;
      cd[bp] = 1.0e30f;
    }
  }
}

// ---------------- GCN aggregate: (m[idx].sum + m)/(k+1) + b [,relu] ---------
__global__ void gcn_agg(const float* __restrict__ M, const int* __restrict__ idx,
                        const float* __restrict__ bias, float* __restrict__ out,
                        int C, int relu) {
  int tid = blockIdx.x * blockDim.x + threadIdx.x;
  if (tid >= NN * C) return;
  int i = tid / C, c = tid - i * C;
  float acc = M[(size_t)i * C + c];
#pragma unroll
  for (int q = 0; q < KNN; ++q) acc += M[(size_t)idx[i * KNN + q] * C + c];
  float r = acc * (1.0f / (KNN + 1)) + bias[c];
  if (relu) r = fmaxf(r, 0.f);
  out[(size_t)i * C + c] = r;
}

__global__ void matvec(const float* __restrict__ H1, const float* __restrict__ W2,
                       float* __restrict__ m2) {
  int i = blockIdx.x * blockDim.x + threadIdx.x;
  if (i >= NN) return;
  float acc = 0.f;
  for (int c = 0; c < CC; ++c) acc = fmaf(H1[(size_t)i * CC + c], W2[c], acc);
  m2[i] = acc;
}

// ---------------- launcher --------------------------------------------------
extern "C" void kernel_launch(void* const* d_in, const int* in_sizes, int n_in,
                              void* d_out, int out_size, void* d_ws, size_t ws_size,
                              hipStream_t stream) {
  const float* x   = (const float*)d_in[0];
  const float* Wq  = (const float*)d_in[1];
  const float* bq  = (const float*)d_in[2];
  const float* Wk  = (const float*)d_in[3];
  const float* bk  = (const float*)d_in[4];
  const float* Wv  = (const float*)d_in[5];
  const float* bv  = (const float*)d_in[6];
  const float* Wsk = (const float*)d_in[7];
  const float* bsk = (const float*)d_in[8];
  const float* W1  = (const float*)d_in[9];
  const float* b1  = (const float*)d_in[10];
  const float* W2  = (const float*)d_in[11];
  const float* b2  = (const float*)d_in[12];
  float* out = (float*)d_out;

  const size_t NH = (size_t)NN * CC;
  __bf16* qbf = (__bf16*)d_ws;          // [NN][CC] bf16
  __bf16* kbf = qbf + NH;               // [NN][CC] bf16
  __bf16* vtb = kbf + NH;               // [CC][NN] bf16 (transposed V)
  float*  sk  = (float*)(vtb + NH);     // skip projection fp32
  float*  h   = sk + NH;
  float*  mt  = h  + NH;
  float*  h1  = mt + NH;
  float*  sq  = h1 + NH;
  float*  m2  = sq + NN;
  int*    idx = (int*)(m2 + NN);

  dim3 gg(NN / 16, CC / 16);
  gemm_bias<<<gg, 32, 0, stream>>>(x, Wq,  bq,  nullptr, qbf,    nullptr, CC, CC);
  gemm_bias<<<gg, 32, 0, stream>>>(x, Wk,  bk,  nullptr, kbf,    nullptr, CC, CC);
  gemm_bias<<<gg, 32, 0, stream>>>(x, Wv,  bv,  nullptr, nullptr, vtb,    CC, CC);
  gemm_bias<<<gg, 32, 0, stream>>>(x, Wsk, bsk, sk,      nullptr, nullptr, CC, CC);

  attn<<<NN / 16, 32, 0, stream>>>(qbf, kbf, vtb, sk, h);

  sqnorm<<<NN / 256, 256, 0, stream>>>(h, sq);
  knn<<<NN, 128, 0, stream>>>(h, sq, idx);

  gemm_bias<<<gg, 32, 0, stream>>>(h, W1, nullptr, mt, nullptr, nullptr, CC, CC);
  gcn_agg<<<(NN * CC + 255) / 256, 256, 0, stream>>>(mt, idx, b1, h1, CC, 1);

  matvec<<<(NN + 255) / 256, 256, 0, stream>>>(h1, W2, m2);
  gcn_agg<<<(NN + 255) / 256, 256, 0, stream>>>(m2, idx, b2, out, 1, 0);
}
